// MultiHeadedSelfAttention_75299366633858
// MI455X (gfx1250) — matrix operand order repaired and checked by reading, not compiled, not run on hardware
//
#include <hip/hip_runtime.h>

// ---------------------------------------------------------------------------
// MI455X (gfx1250) fused multi-head self-attention with double softmax.
//   Kernel 0: fp32 -> bf16 conversion of x and W (bandwidth-trivial pre-pass)
//   Kernel 1: QKV projection GEMM, bf16 WMMA, async-to-LDS double-buffered
//             staging; Q/K stored head-major, V stored TRANSPOSED [b][h][w][s]
//   Kernel 2: streaming attention, two passes over keys (recompute scores),
//             no S x S materialization, no LDS barriers in the hot loops.
// All matmuls via v_wmma_f32_16x16x32_bf16 (wave32).
// ---------------------------------------------------------------------------

typedef __attribute__((ext_vector_type(16))) __bf16 v16bf;
typedef __attribute__((ext_vector_type(8)))  float  v8f;
typedef __attribute__((ext_vector_type(4)))  int    v4i;

#define DMODEL 1024
#define SEQ    2048
#define NB     2
#define NH     16
#define HW     64
#define BS     (NB * SEQ)   // 4096 rows

// workspace layout in bf16 units
#define WS_QKV 0                         // 3 * BS * DMODEL
#define WS_XBF (3 * BS * DMODEL)         // BS * DMODEL
#define WS_WBF (WS_XBF + BS * DMODEL)    // 3 * DMODEL * DMODEL

// CDNA5 async global->LDS copy path (feature-guarded; sync fallback otherwise)
#if defined(__HIP_DEVICE_COMPILE__) && \
    __has_builtin(__builtin_amdgcn_global_load_async_to_lds_b128) && \
    __has_builtin(__builtin_amdgcn_s_wait_asynccnt)
#define ASYNC_LDS 1
// builtin signature (from hipcc diagnostic): (int4 AS1*, int4 AS3*, imm, imm)
#define GV4(p) ((__attribute__((address_space(1))) v4i*)(p))
#define LV4(p) ((__attribute__((address_space(3))) v4i*)(p))
#endif

union Frag {
  uint4 u[2];   // two 16B chunks per lane
  v16bf v;      // 16 bf16 elements
};

__device__ __forceinline__ unsigned short f2bf(float f) {
  unsigned int u = __float_as_uint(f);
  u += 0x7fffu + ((u >> 16) & 1u);          // round-to-nearest-even
  return (unsigned short)(u >> 16);
}

__device__ __forceinline__ v8f vzero8() {
  v8f z;
#pragma unroll
  for (int i = 0; i < 8; ++i) z[i] = 0.0f;
  return z;
}

// ---------------------------------------------------------------------------
// Kernel 0: fp32 -> bf16, 4 elements/thread, fully coalesced.
// ---------------------------------------------------------------------------
__global__ __launch_bounds__(256) void cvt_bf16_kernel(
    const float* __restrict__ src, unsigned short* __restrict__ dst, int n4) {
  int i = blockIdx.x * 256 + threadIdx.x;
  if (i < n4) {
    float4 f = ((const float4*)src)[i];
    unsigned int p0 = (unsigned int)f2bf(f.x) | ((unsigned int)f2bf(f.y) << 16);
    unsigned int p1 = (unsigned int)f2bf(f.z) | ((unsigned int)f2bf(f.w) << 16);
    ((uint2*)dst)[i] = make_uint2(p0, p1);
  }
}

// ---------------------------------------------------------------------------
// Kernel 1: y = x @ W.T + b for W in {Wq, Wk, Wv} (blockIdx.z selects).
// Block tile 128(M) x 128(N); K staged 32-wide in double-buffered bf16 LDS
// via global_load_async_to_lds_b128 (ASYNCcnt), overlapped with WMMA.
// 8 waves arranged 2(M) x 4(N); each wave owns a 64x32 C tile (4x2 accs).
// z<2 : dst[((b*NH + h)*SEQ + s)*HW + w]          (Q, K head-major)
// z==2: dst[((b*NH + h)*HW  + w)*SEQ + s]         (V transposed, b128 stores)
// ---------------------------------------------------------------------------
__global__ __launch_bounds__(256) void qkv_gemm_kernel(
    const unsigned short* __restrict__ xbf,
    const unsigned short* __restrict__ wbf,
    const float* __restrict__ bq, const float* __restrict__ bk,
    const float* __restrict__ bv,
    unsigned short* __restrict__ qkv) {
  const int z = blockIdx.z;
  const unsigned short* __restrict__ Wp = wbf + (size_t)z * DMODEL * DMODEL;
  const float* __restrict__ bp = (z == 0) ? bq : (z == 1) ? bk : bv;
  unsigned short* __restrict__ dst = qkv + (size_t)z * BS * DMODEL;

  // stride 56 shorts = 112B: 16B-aligned rows, 28-bank step -> conflict-free b128
  __shared__ __align__(16) unsigned short Xs[2][128][56];
  __shared__ __align__(16) unsigned short Wt[2][128][56];

  const int tid  = threadIdx.x;
  const int lane = tid & 31;
  const int wid  = tid >> 5;
  const int l16  = lane & 15;
  const int half = lane >> 4;
  const int wm   = wid >> 2;        // 0..1
  const int wn   = wid & 3;         // 0..3
  const int m0   = blockIdx.y * 128;
  const int n0   = blockIdx.x * 128;

  v8f acc[4][2];
#pragma unroll
  for (int ni = 0; ni < 2; ++ni) {
    float bb = bp[n0 + wn * 32 + ni * 16 + l16];   // bias depends on col only
#pragma unroll
    for (int mi = 0; mi < 4; ++mi)
#pragma unroll
      for (int r = 0; r < 8; ++r) acc[mi][ni][r] = bb;
  }

  // stage one 128x32 bf16 tile pair: 512 16B chunks each, 2 per thread each
  auto stage = [&](int k0, int buf) {
#pragma unroll
    for (int j = 0; j < 2; ++j) {
      int c    = 2 * tid + j;          // 0..511
      int row  = c >> 2;               // 0..127
      int coff = (c & 3) * 8;          // bf16 entries, 16B chunks
      const unsigned short* gx = &xbf[(size_t)(m0 + row) * DMODEL + k0 + coff];
      const unsigned short* gw = &Wp [(size_t)(n0 + row) * DMODEL + k0 + coff];
#ifdef ASYNC_LDS
      __builtin_amdgcn_global_load_async_to_lds_b128(
          GV4(gx), LV4(&Xs[buf][row][coff]), 0, 0);
      __builtin_amdgcn_global_load_async_to_lds_b128(
          GV4(gw), LV4(&Wt[buf][row][coff]), 0, 0);
#else
      *(uint4*)&Xs[buf][row][coff] = *(const uint4*)gx;
      *(uint4*)&Wt[buf][row][coff] = *(const uint4*)gw;
#endif
    }
  };

  stage(0, 0);
#ifdef ASYNC_LDS
  __builtin_amdgcn_s_wait_asynccnt(0);
#endif
  __syncthreads();

  for (int k0 = 0; k0 < DMODEL; k0 += 32) {
    const int buf = (k0 >> 5) & 1;
    if (k0 + 32 < DMODEL) stage(k0 + 32, buf ^ 1);   // overlap copy with WMMA

    // A fragments: lane = row M, K chunks {half*8.., 16+half*8..}
    Frag a[4], b[2];
#pragma unroll
    for (int mi = 0; mi < 4; ++mi) {
      int rm = wm * 64 + mi * 16 + l16;
      a[mi].u[0] = *(const uint4*)&Xs[buf][rm][half * 8];
      a[mi].u[1] = *(const uint4*)&Xs[buf][rm][16 + half * 8];
    }
    // B fragments: lane = col N, K chunks {half*8.., 16+half*8..}  (B[k][n] = W[n][k])
#pragma unroll
    for (int ni = 0; ni < 2; ++ni) {
      int rn = wn * 32 + ni * 16 + l16;
      b[ni].u[0] = *(const uint4*)&Wt[buf][rn][half * 8];
      b[ni].u[1] = *(const uint4*)&Wt[buf][rn][16 + half * 8];
    }
#pragma unroll
    for (int mi = 0; mi < 4; ++mi)
#pragma unroll
      for (int ni = 0; ni < 2; ++ni)
        acc[mi][ni] = __builtin_amdgcn_wmma_f32_16x16x32_bf16(
            false, a[mi].v, false, b[ni].v, (short)0, acc[mi][ni], false, false);

#ifdef ASYNC_LDS
    __builtin_amdgcn_s_wait_asynccnt(0);   // next tile landed in LDS
#endif
    __syncthreads();                       // all waves done reading this tile
  }

  // C layout: M = r + 8*half (consecutive r -> consecutive s), N = l16.
  if (z == 2) {
    // V transposed: per (mi,ni) tile a lane owns 8 consecutive s -> one b128
#pragma unroll
    for (int mi = 0; mi < 4; ++mi) {
#pragma unroll
      for (int ni = 0; ni < 2; ++ni) {
        int gn = n0 + wn * 32 + ni * 16 + l16;
        int hh = gn >> 6, w = gn & 63;
        int gs = m0 + wm * 64 + mi * 16 + 8 * half;   // first of 8 s values
        int bb = gs >> 11, s = gs & 2047;
        uint4 pk;
        pk.x = (unsigned int)f2bf(acc[mi][ni][0]) | ((unsigned int)f2bf(acc[mi][ni][1]) << 16);
        pk.y = (unsigned int)f2bf(acc[mi][ni][2]) | ((unsigned int)f2bf(acc[mi][ni][3]) << 16);
        pk.z = (unsigned int)f2bf(acc[mi][ni][4]) | ((unsigned int)f2bf(acc[mi][ni][5]) << 16);
        pk.w = (unsigned int)f2bf(acc[mi][ni][6]) | ((unsigned int)f2bf(acc[mi][ni][7]) << 16);
        *(uint4*)&dst[((size_t)(bb * NH + hh) * HW + w) * SEQ + s] = pk;
      }
    }
  } else {
#pragma unroll
    for (int mi = 0; mi < 4; ++mi) {
#pragma unroll
      for (int ni = 0; ni < 2; ++ni) {
        int gn = n0 + wn * 32 + ni * 16 + l16;
        int hh = gn >> 6, w = gn & 63;
#pragma unroll
        for (int r = 0; r < 8; ++r) {
          int gm = m0 + wm * 64 + mi * 16 + r + 8 * half;
          int bb = gm >> 11, s = gm & 2047;
          dst[((size_t)(bb * NH + hh) * SEQ + s) * HW + w] = f2bf(acc[mi][ni][r]);
        }
      }
    }
  }
}

// ---------------------------------------------------------------------------
// Kernel 2: streaming attention with double softmax.
//   softmax(softmax(s)) @ V  ==  sum_j exp(p_j) v_j / sum_j exp(p_j),
//   p_j = exp(s_j - m1)/Z1.  Pass A: m1,Z1.  Pass B: recompute s, accumulate.
// One wave per 16-query tile; 8 waves/block share one (b,h) so K/V stay in
// WGP$/L2 (512 KB per head).  V is pre-transposed, so pass B needs no LDS
// staging and no barriers at all.
// ---------------------------------------------------------------------------
__global__ __launch_bounds__(256) void attn_kernel(
    const unsigned short* __restrict__ qb,
    const unsigned short* __restrict__ kb,
    const unsigned short* __restrict__ vtb,
    float* __restrict__ out) {
  const int tid  = threadIdx.x;
  const int lane = tid & 31;
  const int wid  = tid >> 5;
  const int l16  = lane & 15;
  const int half = lane >> 4;
  const int bh   = blockIdx.x >> 4;       // 0..31
  const int qblk = blockIdx.x & 15;
  const int b    = bh >> 4, h = bh & 15;
  const int q0   = (qblk * 8 + wid) * 16;

  const unsigned short* __restrict__ qh = qb  + (size_t)bh * SEQ * HW;
  const unsigned short* __restrict__ kh = kb  + (size_t)bh * SEQ * HW;
  const unsigned short* __restrict__ vt = vtb + (size_t)bh * HW * SEQ;  // [w][s]

  // Pt: per-wave probability tile [query m][key], A-layout friendly.
  // 72-short stride: 16B-aligned rows, 36-bank step -> conflict-free b128.
  __shared__ __align__(16) unsigned short Pt[8][16][72];

  // Q fragments (16 x 64 bf16), k-steps of 32
  Frag aq[2];
#pragma unroll
  for (int ks = 0; ks < 2; ++ks) {
    const unsigned short* qrow = &qh[(size_t)(q0 + l16) * HW + ks * 32];
    aq[ks].u[0] = *(const uint4*)&qrow[half * 8];
    aq[ks].u[1] = *(const uint4*)&qrow[16 + half * 8];
  }

  float m1[8], Z1[8];
#pragma unroll
  for (int r = 0; r < 8; ++r) { m1[r] = -1e30f; Z1[r] = 0.0f; }

  // ---------------- pass A: row max + first-softmax denominator -------------
  for (int k0 = 0; k0 < SEQ; k0 += 64) {
    v8f sc[4];
#pragma unroll
    for (int nt = 0; nt < 4; ++nt) {
      sc[nt] = vzero8();
#pragma unroll
      for (int ks = 0; ks < 2; ++ks) {
        Frag bk_;  // B[k=w][n=key] = K[key][w]: two 16B chunks per lane
        const unsigned short* kr =
            &kh[(size_t)(k0 + nt * 16 + l16) * HW + ks * 32];
        bk_.u[0] = *(const uint4*)&kr[half * 8];
        bk_.u[1] = *(const uint4*)&kr[16 + half * 8];
        sc[nt] = __builtin_amdgcn_wmma_f32_16x16x32_bf16(
            false, aq[ks].v, false, bk_.v, (short)0, sc[nt], false, false);
      }
    }
#pragma unroll
    for (int r = 0; r < 8; ++r) {
      float mx = fmaxf(fmaxf(sc[0][r], sc[1][r]), fmaxf(sc[2][r], sc[3][r])) * 0.125f;
#pragma unroll
      for (int off = 1; off < 16; off <<= 1) mx = fmaxf(mx, __shfl_xor(mx, off, 32));
      float nm = fmaxf(m1[r], mx);
      float sm = __expf(sc[0][r] * 0.125f - nm) + __expf(sc[1][r] * 0.125f - nm)
               + __expf(sc[2][r] * 0.125f - nm) + __expf(sc[3][r] * 0.125f - nm);
#pragma unroll
      for (int off = 1; off < 16; off <<= 1) sm += __shfl_xor(sm, off, 32);
      Z1[r] = Z1[r] * __expf(m1[r] - nm) + sm;
      m1[r] = nm;
    }
  }
  float iZ1[8];
#pragma unroll
  for (int r = 0; r < 8; ++r) iZ1[r] = 1.0f / Z1[r];

  // ---------------- pass B: recompute scores, accumulate H ------------------
  v8f ho[4];
#pragma unroll
  for (int wt = 0; wt < 4; ++wt) ho[wt] = vzero8();
  float z2[8];
#pragma unroll
  for (int r = 0; r < 8; ++r) z2[r] = 0.0f;

  for (int k0 = 0; k0 < SEQ; k0 += 64) {
    v8f sc[4];
#pragma unroll
    for (int nt = 0; nt < 4; ++nt) {
      sc[nt] = vzero8();
#pragma unroll
      for (int ks = 0; ks < 2; ++ks) {
        Frag bk_;
        const unsigned short* kr =
            &kh[(size_t)(k0 + nt * 16 + l16) * HW + ks * 32];
        bk_.u[0] = *(const uint4*)&kr[half * 8];
        bk_.u[1] = *(const uint4*)&kr[16 + half * 8];
        sc[nt] = __builtin_amdgcn_wmma_f32_16x16x32_bf16(
            false, aq[ks].v, false, bk_.v, (short)0, sc[nt], false, false);
      }
    }

    // w = exp(p) with p = exp(s - m1)/Z1; stash bf16 in A-layout LDS tile
#pragma unroll
    for (int nt = 0; nt < 4; ++nt)
#pragma unroll
      for (int r = 0; r < 8; ++r) {
        float p  = __expf(sc[nt][r] * 0.125f - m1[r]) * iZ1[r];
        float we = __expf(p);                    // p in [0,1], no shift needed
        z2[r] += we;                              // per-lane partial of Z2
        Pt[wid][r + 8 * half][nt * 16 + l16] = f2bf(we);
      }

    // H(16x64) += P(16x64) @ Vblk(64x64); in-wave LDS ordering is in-order,
    // V^T B-fragments come straight from global (two 16B chunks per lane).
#pragma unroll
    for (int ks = 0; ks < 2; ++ks) {
      Frag ap;
      ap.u[0] = *(const uint4*)&Pt[wid][l16][ks * 32 + half * 8];
      ap.u[1] = *(const uint4*)&Pt[wid][l16][ks * 32 + 16 + half * 8];
#pragma unroll
      for (int wt = 0; wt < 4; ++wt) {
        Frag bv_;   // B[k=key][n=w] = Vt[w][key]
        const unsigned short* vr =
            &vt[(size_t)(wt * 16 + l16) * SEQ + k0 + ks * 32];
        bv_.u[0] = *(const uint4*)&vr[half * 8];
        bv_.u[1] = *(const uint4*)&vr[16 + half * 8];
        ho[wt] = __builtin_amdgcn_wmma_f32_16x16x32_bf16(
            false, ap.v, false, bv_.v, (short)0, ho[wt], false, false);
      }
    }
  }

  // finish Z2 across the 16-lane half group, normalize, store (B,S,D) fp32
#pragma unroll
  for (int r = 0; r < 8; ++r) {
#pragma unroll
    for (int off = 1; off < 16; off <<= 1) z2[r] += __shfl_xor(z2[r], off, 32);
  }
#pragma unroll
  for (int wt = 0; wt < 4; ++wt) {
    int w = wt * 16 + l16;
#pragma unroll
    for (int r = 0; r < 8; ++r) {
      int srow = q0 + r + 8 * half;
      out[(size_t)(b * SEQ + srow) * DMODEL + h * HW + w] = ho[wt][r] / z2[r];
    }
  }
}

// ---------------------------------------------------------------------------
extern "C" void kernel_launch(void* const* d_in, const int* in_sizes, int n_in,
                              void* d_out, int out_size, void* d_ws, size_t ws_size,
                              hipStream_t stream) {
  const float* x  = (const float*)d_in[0];
  const float* Wq = (const float*)d_in[1];
  const float* bq = (const float*)d_in[2];
  const float* Wk = (const float*)d_in[3];
  const float* bk = (const float*)d_in[4];
  const float* Wv = (const float*)d_in[5];
  const float* bv = (const float*)d_in[6];
  float* out = (float*)d_out;

  unsigned short* ws  = (unsigned short*)d_ws;
  unsigned short* qkv = ws + WS_QKV;   // q,k head-major; v transposed
  unsigned short* xbf = ws + WS_XBF;
  unsigned short* wbf = ws + WS_WBF;

  // fp32 -> bf16 pre-pass (x, Wq, Wk, Wv)
  cvt_bf16_kernel<<<(BS * DMODEL / 4 + 255) / 256, 256, 0, stream>>>(
      x, xbf, BS * DMODEL / 4);
  cvt_bf16_kernel<<<(DMODEL * DMODEL / 4 + 255) / 256, 256, 0, stream>>>(
      Wq, wbf + 0 * (size_t)DMODEL * DMODEL, DMODEL * DMODEL / 4);
  cvt_bf16_kernel<<<(DMODEL * DMODEL / 4 + 255) / 256, 256, 0, stream>>>(
      Wk, wbf + 1 * (size_t)DMODEL * DMODEL, DMODEL * DMODEL / 4);
  cvt_bf16_kernel<<<(DMODEL * DMODEL / 4 + 255) / 256, 256, 0, stream>>>(
      Wv, wbf + 2 * (size_t)DMODEL * DMODEL, DMODEL * DMODEL / 4);

  // QKV projections: grid (N/128, M/128, 3)
  qkv_gemm_kernel<<<dim3(DMODEL / 128, BS / 128, 3), 256, 0, stream>>>(
      xbf, wbf, bq, bk, bv, qkv);

  // Attention: 32 (b,h) pairs x 16 blocks, 8 query-tiles (waves) per block
  attn_kernel<<<dim3(32 * 16), 256, 0, stream>>>(
      qkv, qkv + (size_t)BS * DMODEL, qkv + 2 * (size_t)BS * DMODEL, out);
}
